// InstancewiseGNN_71614284693721
// MI455X (gfx1250) — matrix operand order repaired and checked by reading, not compiled, not run on hardware
//
#include <hip/hip_runtime.h>
#include <math.h>

#define NNODES 100000
#define NEDGES 1600000
#define HID 128
#define NCLS 40
#define NCLS_PAD 48
#define DEPTHS 4

typedef __attribute__((ext_vector_type(16))) _Float16 v16h;
typedef __attribute__((ext_vector_type(8)))  float    v8f;

// ---------------------------------------------------------------- utilities

__global__ void k_fill(float* __restrict__ p, int n, float v) {
    int i = blockIdx.x * blockDim.x + threadIdx.x;
    if (i < n) p[i] = v;
}

__global__ void k_degree(const int* __restrict__ dst, float* __restrict__ deg, int nE) {
    int e = blockIdx.x * blockDim.x + threadIdx.x;
    if (e < nE) atomicAdd(&deg[dst[e]], 1.0f);
}

// Transpose + convert weight W[K,Nn] (row-major f32) -> Bt[Npad][K] f16 (row = output col)
__global__ void k_prep_w(const float* __restrict__ W, _Float16* __restrict__ Bt,
                         int K, int Nn, int Npad) {
    int idx = blockIdx.x * blockDim.x + threadIdx.x;
    if (idx >= Npad * K) return;
    int n = idx / K, k = idx % K;
    Bt[(size_t)n * K + k] = (n < Nn) ? (_Float16)W[(size_t)k * Nn + n] : (_Float16)0.0f;
}

// ------------------------------------------------------------ WMMA GEMM
// C[M, ldc(<=NT*16)] = A[M,128] * Bt^T + bias ; wave computes one 16-row strip,
// NT column tiles of 16. A converted f32->f16 on the fly; K = 128 fixed.
template <int NT>
__global__ void k_gemm(const float* __restrict__ A, const _Float16* __restrict__ Bt,
                       const float* __restrict__ bias, float* __restrict__ C,
                       int Mtiles, int ldc, int nvalid, int doRelu) {
    int wave = (blockIdx.x * blockDim.x + threadIdx.x) >> 5;
    int lane = threadIdx.x & 31;
    if (wave >= Mtiles) return;
    int m0   = wave * 16;
    int mrow = m0 + (lane & 15);
    // A fragment K-groups: contiguous K half per lane half (matches B)
    int kgA   = (lane < 16) ? 0 : 16;
    int kgB   = (lane < 16) ? 8 : 24;
    int kOffB = (lane < 16) ? 0 : 16;   // B fragment K half per 32x16 layout
    int ncol  = lane & 15;

    v8f acc[NT] = {};

    for (int kb = 0; kb < 128; kb += 32) {
        const float* ar = A + (size_t)mrow * HID + kb;
        float4 fa0 = *(const float4*)(ar + kgA);
        float4 fa1 = *(const float4*)(ar + kgA + 4);
        float4 fb0 = *(const float4*)(ar + kgB);
        float4 fb1 = *(const float4*)(ar + kgB + 4);
        v16h a;
        a[0] = (_Float16)fa0.x;  a[1] = (_Float16)fa0.y;
        a[2] = (_Float16)fa0.z;  a[3] = (_Float16)fa0.w;
        a[4] = (_Float16)fa1.x;  a[5] = (_Float16)fa1.y;
        a[6] = (_Float16)fa1.z;  a[7] = (_Float16)fa1.w;
        a[8] = (_Float16)fb0.x;  a[9] = (_Float16)fb0.y;
        a[10] = (_Float16)fb0.z; a[11] = (_Float16)fb0.w;
        a[12] = (_Float16)fb1.x; a[13] = (_Float16)fb1.y;
        a[14] = (_Float16)fb1.z; a[15] = (_Float16)fb1.w;
#pragma unroll
        for (int t = 0; t < NT; ++t) {
            v16h b = *(const v16h*)(Bt + (size_t)(t * 16 + ncol) * HID + kb + kOffB);
            acc[t] = __builtin_amdgcn_wmma_f32_16x16x32_f16(
                false, a, false, b, (short)0, acc[t], false, false);
        }
    }

    int mhi = (lane >= 16) ? 8 : 0;
#pragma unroll
    for (int t = 0; t < NT; ++t) {
        int n = t * 16 + ncol;
        float bv = (n < nvalid) ? bias[n] : 0.0f;
#pragma unroll
        for (int r = 0; r < 8; ++r) {
            float v = acc[t][r] + bv;
            if (doRelu) v = fmaxf(v, 0.0f);
            if (n < nvalid) C[(size_t)(m0 + mhi + r) * ldc + n] = v;
        }
    }
}

// --------------------------------------------------- controller (wave/node)
__global__ void k_controller(const float* __restrict__ h,
                             const float* __restrict__ Wd, const float* __restrict__ bd,
                             const float* __restrict__ Wa, const float* __restrict__ ba,
                             float* __restrict__ probs) {
    int wave = (blockIdx.x * blockDim.x + threadIdx.x) >> 5;
    int lane = threadIdx.x & 31;
    if (wave >= NNODES) return;
    float4 hv = ((const float4*)(h + (size_t)wave * HID))[lane];
    float hx[4] = {hv.x, hv.y, hv.z, hv.w};
    float s0 = 0, s1 = 0, a0 = 0, a1 = 0, a2 = 0;
    int k0 = lane * 4;
#pragma unroll
    for (int j = 0; j < 4; ++j) {
        int k = k0 + j;
        s0 += hx[j] * Wd[k * 2 + 0];
        s1 += hx[j] * Wd[k * 2 + 1];
        a0 += hx[j] * Wa[k * 3 + 0];
        a1 += hx[j] * Wa[k * 3 + 1];
        a2 += hx[j] * Wa[k * 3 + 2];
    }
    for (int off = 16; off > 0; off >>= 1) {
        s0 += __shfl_xor(s0, off); s1 += __shfl_xor(s1, off);
        a0 += __shfl_xor(a0, off); a1 += __shfl_xor(a1, off); a2 += __shfl_xor(a2, off);
    }
    if (lane == 0) {
        s0 += bd[0]; s1 += bd[1];
        float m  = fmaxf(s0, s1);
        float e0 = __expf(s0 - m), e1 = __expf(s1 - m);
        float inv = 1.0f / (e0 + e1);
        a0 += ba[0]; a1 += ba[1]; a2 += ba[2];
        float ma = fmaxf(a0, fmaxf(a1, a2));
        float ea0 = __expf(a0 - ma), ea1 = __expf(a1 - ma), ea2 = __expf(a2 - ma);
        float inva = 1.0f / (ea0 + ea1 + ea2);
        float* p = probs + (size_t)wave * 5;
        p[0] = e0 * inv;  p[1] = e1 * inv;
        p[2] = ea0 * inva; p[3] = ea1 * inva; p[4] = ea2 * inva;
    }
}

// ------------------------------------------------ edge aggregation (wave/edge)
__device__ __forceinline__ void atomicMaxF(float* addr, float v) {
    if (v >= 0.0f) atomicMax((int*)addr, __float_as_int(v));
    else           atomicMin((unsigned int*)addr, __float_as_uint(v));
}

__global__ void k_edge(const int* __restrict__ src, const int* __restrict__ dst,
                       const float* __restrict__ h2,
                       float* __restrict__ meanb, float* __restrict__ maxb) {
    int wave = (blockIdx.x * blockDim.x + threadIdx.x) >> 5;
    int lane = threadIdx.x & 31;
    if (wave >= NEDGES) return;
    int s = src[wave], d = dst[wave];
    float4 v = ((const float4*)(h2 + (size_t)s * HID))[lane];
    float* mrow = meanb + (size_t)d * HID + lane * 4;
    float* xrow = maxb  + (size_t)d * HID + lane * 4;
    atomicAdd(mrow + 0, v.x); atomicAdd(mrow + 1, v.y);
    atomicAdd(mrow + 2, v.z); atomicAdd(mrow + 3, v.w);
    atomicMaxF(xrow + 0, v.x); atomicMaxF(xrow + 1, v.y);
    atomicMaxF(xrow + 2, v.z); atomicMaxF(xrow + 3, v.w);
}

// --------------------------------------------------- mix + halt (wave/node)
__global__ void k_combine(const float* __restrict__ h2, const float* __restrict__ meanb,
                          const float* __restrict__ maxb, const float* __restrict__ deg,
                          const float* __restrict__ probs, float* __restrict__ contpr,
                          float* __restrict__ h, float* __restrict__ selected) {
    int wave = (blockIdx.x * blockDim.x + threadIdx.x) >> 5;
    int lane = threadIdx.x & 31;
    if (wave >= NNODES) return;
    const float* p = probs + (size_t)wave * 5;
    float stop0 = p[0], stop1 = p[1];
    float w_mean = p[2], w_max = p[3], w_self = p[4];
    float cp   = contpr[wave];
    float invd = 1.0f / fmaxf(deg[wave], 1.0f);
    float gsc  = cp * stop1;
    size_t off = (size_t)wave * HID + lane * 4;
    float4 hm  = *(const float4*)(meanb + off);
    float4 hx  = *(const float4*)(maxb + off);
    float4 hs  = *(const float4*)(h2 + off);
    float4 sel = *(const float4*)(selected + off);
    float mn[4] = {hm.x, hm.y, hm.z, hm.w};
    float mx[4] = {hx.x, hx.y, hx.z, hx.w};
    float sv[4] = {hs.x, hs.y, hs.z, hs.w};
    float se[4] = {sel.x, sel.y, sel.z, sel.w};
    float o[4];
#pragma unroll
    for (int j = 0; j < 4; ++j) {
        float m = __builtin_isfinite(mx[j]) ? mx[j] : 0.0f;  // isolated nodes -> 0
        float hn = fmaxf(w_mean * mn[j] * invd + w_max * m + w_self * sv[j], 0.0f);
        o[j]  = hn;
        se[j] += gsc * hn;
    }
    *(float4*)(h + off)        = make_float4(o[0], o[1], o[2], o[3]);
    *(float4*)(selected + off) = make_float4(se[0], se[1], se[2], se[3]);
    if (lane == 0) contpr[wave] = cp * stop0;
}

// --------------------------------------------------- log-softmax (wave/node)
__global__ void k_logsoftmax(float* __restrict__ out) {
    int wave = (blockIdx.x * blockDim.x + threadIdx.x) >> 5;
    int lane = threadIdx.x & 31;
    if (wave >= NNODES) return;
    float* row = out + (size_t)wave * NCLS;
    float v0 = (lane < NCLS) ? row[lane] : -INFINITY;
    float v1 = (lane + 32 < NCLS) ? row[lane + 32] : -INFINITY;
    float m = fmaxf(v0, v1);
    for (int off = 16; off > 0; off >>= 1) m = fmaxf(m, __shfl_xor(m, off));
    float e = ((lane < NCLS) ? __expf(v0 - m) : 0.0f) +
              ((lane + 32 < NCLS) ? __expf(v1 - m) : 0.0f);
    for (int off = 16; off > 0; off >>= 1) e += __shfl_xor(e, off);
    float ls = m + __logf(e);
    if (lane < NCLS)      row[lane]      = v0 - ls;
    if (lane + 32 < NCLS) row[lane + 32] = v1 - ls;
}

// ---------------------------------------------------------------- launcher
extern "C" void kernel_launch(void* const* d_in, const int* in_sizes, int n_in,
                              void* d_out, int out_size, void* d_ws, size_t ws_size,
                              hipStream_t stream) {
    const float* x       = (const float*)d_in[0];
    const int*   edge    = (const int*)d_in[1];
    const int*   src     = edge;
    const int*   dst     = edge + NEDGES;
    const float* W_pre   = (const float*)d_in[2];
    const float* b_pre   = (const float*)d_in[3];
    const float* W_depth = (const float*)d_in[4];
    const float* b_depth = (const float*)d_in[5];
    const float* W_aggr  = (const float*)d_in[6];
    const float* b_aggr  = (const float*)d_in[7];
    const float* W_conv  = (const float*)d_in[8];
    const float* b_conv  = (const float*)d_in[9];
    const float* W_post  = (const float*)d_in[10];
    const float* b_post  = (const float*)d_in[11];

    // carve workspace (256B aligned)
    char*  ws = (char*)d_ws;
    auto carve = [&](size_t bytes) -> char* {
        char* p = ws;
        ws += (bytes + 255) & ~(size_t)255;
        return p;
    };
    const size_t NHf = (size_t)NNODES * HID * sizeof(float);
    float*    h        = (float*)carve(NHf);
    float*    h2       = (float*)carve(NHf);
    float*    meanb    = (float*)carve(NHf);
    float*    maxb     = (float*)carve(NHf);
    float*    selected = (float*)carve(NHf);
    float*    probs    = (float*)carve((size_t)NNODES * 5 * sizeof(float));
    float*    deg      = (float*)carve((size_t)NNODES * sizeof(float));
    float*    contpr   = (float*)carve((size_t)NNODES * sizeof(float));
    _Float16* Bt_pre   = (_Float16*)carve((size_t)HID * HID * sizeof(_Float16));
    _Float16* Bt_conv  = (_Float16*)carve((size_t)DEPTHS * HID * HID * sizeof(_Float16));
    _Float16* Bt_post  = (_Float16*)carve((size_t)NCLS_PAD * HID * sizeof(_Float16));
    float*    logits   = (float*)d_out;

    const int NHelem = NNODES * HID;
    auto blk  = [](int n, int t) { return (n + t - 1) / t; };
    const int MT         = NNODES / 16;                 // 6250 M-tiles (exact)
    const int gemmBlocks = blk(MT * 32, 256);
    const int nodeBlocks = blk(NNODES * 32, 256);
    const int edgeBlocks = blk(NEDGES, 256);            // thread/edge kernels
    const int edgeWaveBlocks = (NEDGES * 32) / 256;     // wave/edge kernels

    // weight prep (f32 -> f16, transposed to N-major for B fragments)
    k_prep_w<<<blk(HID * HID, 256), 256, 0, stream>>>(W_pre, Bt_pre, HID, HID, HID);
    for (int i = 0; i < DEPTHS; ++i)
        k_prep_w<<<blk(HID * HID, 256), 256, 0, stream>>>(
            W_conv + (size_t)i * HID * HID, Bt_conv + (size_t)i * HID * HID, HID, HID, HID);
    k_prep_w<<<blk(NCLS_PAD * HID, 256), 256, 0, stream>>>(W_post, Bt_post, HID, NCLS, NCLS_PAD);

    // state init
    k_fill<<<blk(NNODES, 256), 256, 0, stream>>>(deg, NNODES, 0.0f);
    k_degree<<<edgeBlocks, 256, 0, stream>>>(dst, deg, NEDGES);
    k_fill<<<blk(NHelem, 256), 256, 0, stream>>>(selected, NHelem, 0.0f);
    k_fill<<<blk(NNODES, 256), 256, 0, stream>>>(contpr, NNODES, 1.0f);

    // pre_mp: h = relu(x @ W_pre + b_pre)
    k_gemm<8><<<gemmBlocks, 256, 0, stream>>>(x, Bt_pre, b_pre, h, MT, HID, HID, 1);

    for (int i = 0; i < DEPTHS; ++i) {
        k_controller<<<nodeBlocks, 256, 0, stream>>>(h, W_depth, b_depth, W_aggr, b_aggr, probs);
        k_gemm<8><<<gemmBlocks, 256, 0, stream>>>(
            h, Bt_conv + (size_t)i * HID * HID, b_conv + (size_t)i * HID, h2, MT, HID, HID, 0);
        k_fill<<<blk(NHelem, 256), 256, 0, stream>>>(meanb, NHelem, 0.0f);
        k_fill<<<blk(NHelem, 256), 256, 0, stream>>>(maxb, NHelem, -INFINITY);
        k_edge<<<edgeWaveBlocks, 256, 0, stream>>>(src, dst, h2, meanb, maxb);
        k_combine<<<nodeBlocks, 256, 0, stream>>>(h2, meanb, maxb, deg, probs, contpr, h, selected);
    }

    // post_mp: logits = selected @ W_post + b_post  (3 N-tiles cover 48, store 40)
    k_gemm<3><<<gemmBlocks, 256, 0, stream>>>(selected, Bt_post, b_post, logits, MT, NCLS, NCLS, 0);
    k_logsoftmax<<<nodeBlocks, 256, 0, stream>>>(logits);
}
